// SceneCentricRelative_8383776162281
// MI455X (gfx1250) — matrix operand order repaired and checked by reading, not compiled, not run on hardware
//
#include <hip/hip_runtime.h>
#include <hip/hip_bf16.h>
#include <math.h>

// ---------------- problem constants ----------------
#define HD      256          // hidden H
#define NHEAD   8
#define DHEAD   32
#define KNN     36
#define NSCENE  4
#define NTOT    1344         // 1024 + 256 + 64
#define TILE_K  32
#define BPAD    40           // padded Bst row (halves); 80B keeps 16B/8B alignment

typedef _Float16 half_t;
typedef half_t v16h __attribute__((ext_vector_type(16)));
typedef half_t v8h  __attribute__((ext_vector_type(8)));
typedef half_t v4h  __attribute__((ext_vector_type(4)));
typedef float  v8f  __attribute__((ext_vector_type(8)));

// =====================================================================
// KNN: one thread per (scene, query). Masked insertion top-K over N keys.
// =====================================================================
__global__ void knn_kernel(const unsigned char* __restrict__ valid,
                           const float* __restrict__ rel_pose,
                           const float* __restrict__ rel_dist,
                           int* __restrict__ idx,
                           unsigned char* __restrict__ invk,
                           float* __restrict__ rpe,
                           int N, int off)
{
    int t = blockIdx.x * blockDim.x + threadIdx.x;
    if (t >= NSCENE * N) return;
    int s = t / N, n = t % N;
    const float*         drow = rel_dist + ((size_t)s * NTOT + off + n) * NTOT + off;
    const unsigned char* vrow = valid + (size_t)s * N;

    float bd[KNN]; int bi[KNN];
#pragma unroll
    for (int k = 0; k < KNN; ++k) { bd[k] = 3.0e38f; bi[k] = 0; }

    for (int j = 0; j < N; ++j) {
        float dj = vrow[j] ? drow[j] : 1.0e9f;
        if (dj < bd[KNN - 1]) {
            int k = KNN - 1;
            while (k > 0 && bd[k - 1] > dj) { bd[k] = bd[k - 1]; bi[k] = bi[k - 1]; --k; }
            bd[k] = dj; bi[k] = j;
        }
    }
    for (int k = 0; k < KNN; ++k) {
        int j = bi[k];
        size_t o = (size_t)t * KNN + k;
        idx[o]  = j;
        invk[o] = (unsigned char)((!vrow[j]) || (bd[k] > 2000.0f));
        const float* p = rel_pose + (((size_t)s * NTOT + off + n) * NTOT + off + j) * 3;
        float th = p[2];
        rpe[o * 4 + 0] = p[0];
        rpe[o * 4 + 1] = p[1];
        rpe[o * 4 + 2] = __cosf(th);
        rpe[o * 4 + 3] = __sinf(th);
    }
}

// =====================================================================
// WMMA GEMM: C[M,Nd] = act(A[M,Kd] x W[Kd,Nd] + bias [+residual]),
// optional row-zeroing by validity mask. f16 inputs, f32 accumulate.
// Block: 128 threads = 4 waves; block tile 64(M) x 32(N); each wave owns
// 16x32 (one A fragment, two B fragments / two accumulators).
// A fragment  : lane(row) = one contiguous 32B run (K=16h)
// B fragment  : Bst stored [n][k] -> one contiguous 32B run (K=16h)
// =====================================================================
__global__ __launch_bounds__(128) void gemm_kernel(
    const float* __restrict__ A, const float* __restrict__ W,
    const float* __restrict__ bias, const float* __restrict__ residual,
    const unsigned char* __restrict__ maskvalid,
    float* __restrict__ C, int M, int Kd, int Nd, int relu)
{
    const int wave  = threadIdx.x >> 5;
    const int lane  = threadIdx.x & 31;
    const int half  = lane >> 4;
    const int row16 = lane & 15;
    const int m0    = blockIdx.x * 64;
    const int n0    = blockIdx.y * 32;

    __shared__ __align__(16) half_t As[64][TILE_K];   // 4 KB
    __shared__ __align__(16) half_t Bst[32][BPAD];    // [n][k], 2.5 KB

    v8f acc0 = {}, acc1 = {};

    for (int k0 = 0; k0 < Kd; k0 += TILE_K) {
        const bool full_k = (k0 + TILE_K <= Kd);
        const bool full   = full_k && (m0 + 64 <= M);
        if (full) {
            // vectorized A stage: 512 float4 loads -> packed 4-half LDS stores
            for (int i = threadIdx.x; i < 512; i += 128) {
                int r = i >> 3, c4 = (i & 7) << 2;
                const float4 f = *(const float4*)&A[(size_t)(m0 + r) * Kd + k0 + c4];
                v4h h = { (half_t)f.x, (half_t)f.y, (half_t)f.z, (half_t)f.w };
                *(v4h*)&As[r][c4] = h;
                if (k0 + TILE_K < Kd)   // uniform branch
                    __builtin_prefetch(&A[(size_t)(m0 + r) * Kd + k0 + c4 + TILE_K], 0, 3);
            }
        } else {
            for (int i = threadIdx.x; i < 64 * TILE_K; i += 128) {
                int r = i >> 5, c = i & 31;
                int gk = k0 + c, gm = m0 + r;
                float v = (gk < Kd && gm < M) ? A[(size_t)gm * Kd + gk] : 0.0f;
                As[r][c] = (half_t)v;
            }
        }
        // B stage column-major: thread owns (n, 4 consecutive k) -> 4 coalesced
        // global b32 reads + one 8-byte ds_store_b64 at Bst[n][4*kg]
        if (full_k) {
            for (int i = threadIdx.x; i < 256; i += 128) {
                int n = i & 31, kbase = (i >> 5) << 2;
                v4h h;
#pragma unroll
                for (int j = 0; j < 4; ++j)
                    h[j] = (half_t)W[(size_t)(k0 + kbase + j) * Nd + (n0 + n)];
                *(v4h*)&Bst[n][kbase] = h;
            }
        } else {
            for (int i = threadIdx.x; i < 256; i += 128) {
                int n = i & 31, kbase = (i >> 5) << 2;
                v4h h;
#pragma unroll
                for (int j = 0; j < 4; ++j) {
                    int gk = k0 + kbase + j;
                    h[j] = (half_t)((gk < Kd) ? W[(size_t)gk * Nd + (n0 + n)] : 0.0f);
                }
                *(v4h*)&Bst[n][kbase] = h;
            }
        }
        __syncthreads();

        // fragments: contiguous 16B LDS loads, k = 16*half + s for both A and B
        const v8h a_lo = *(const v8h*)&As[wave * 16 + row16][16 * half];
        const v8h a_hi = *(const v8h*)&As[wave * 16 + row16][16 * half + 8];
        v16h a = __builtin_shufflevector(a_lo, a_hi,
                    0,1,2,3,4,5,6,7,8,9,10,11,12,13,14,15);
        const v8h b0l = *(const v8h*)&Bst[row16][16 * half];
        const v8h b0h = *(const v8h*)&Bst[row16][16 * half + 8];
        v16h b0 = __builtin_shufflevector(b0l, b0h,
                    0,1,2,3,4,5,6,7,8,9,10,11,12,13,14,15);
        const v8h b1l = *(const v8h*)&Bst[16 + row16][16 * half];
        const v8h b1h = *(const v8h*)&Bst[16 + row16][16 * half + 8];
        v16h b1 = __builtin_shufflevector(b1l, b1h,
                    0,1,2,3,4,5,6,7,8,9,10,11,12,13,14,15);

        acc0 = __builtin_amdgcn_wmma_f32_16x16x32_f16(
                   false, a, false, b0, (short)0, acc0, false, false);
        acc1 = __builtin_amdgcn_wmma_f32_16x16x32_f16(
                   false, a, false, b1, (short)0, acc1, false, false);
        __syncthreads();
    }

    // writeback: D layout: VGPR r -> M = r + 8*half, N = lane%16 (per tile)
    const float bia0 = bias ? bias[n0 + row16]      : 0.0f;
    const float bia1 = bias ? bias[n0 + 16 + row16] : 0.0f;
#pragma unroll
    for (int r = 0; r < 8; ++r) {
        int gm = m0 + wave * 16 + r + half * 8;
        if (gm < M) {
            bool dead = maskvalid && !maskvalid[gm];
            int gn0 = n0 + row16, gn1 = n0 + 16 + row16;
            float v0 = acc0[r] + bia0;
            float v1 = acc1[r] + bia1;
            if (residual) {
                v0 += residual[(size_t)gm * Nd + gn0];
                v1 += residual[(size_t)gm * Nd + gn1];
            }
            if (relu) { v0 = fmaxf(v0, 0.0f); v1 = fmaxf(v1, 0.0f); }
            if (dead) { v0 = 0.0f; v1 = 0.0f; }
            C[(size_t)gm * Nd + gn0] = v0;
            C[(size_t)gm * Nd + gn1] = v1;
        }
    }
}

// =====================================================================
// KNN attention, one block per (s,n); thread = (head h, dim d), DH == wave.
// kk = Kx[idx] + rpe @ Wrk + brk  (rank-4 RPE fused; weights hoisted)
// =====================================================================
__global__ __launch_bounds__(256) void attn_kernel(
    const float* __restrict__ Q,  const float* __restrict__ Kx,
    const float* __restrict__ Vx, const float* __restrict__ rpe,
    const int* __restrict__ idx,  const unsigned char* __restrict__ invk,
    const float* __restrict__ Wrk, const float* __restrict__ brk,
    const float* __restrict__ Wrv, const float* __restrict__ brv,
    float* __restrict__ O, int N)
{
    int sn = blockIdx.x;
    int h  = threadIdx.x >> 5;
    int d  = threadIdx.x & 31;
    int c  = h * DHEAD + d;
    int s  = sn / N;

    float q = Q[(size_t)sn * HD + c];
    __shared__ float sc[NHEAD][KNN];

    const int*           idxr = idx  + (size_t)sn * KNN;
    const unsigned char* ivr  = invk + (size_t)sn * KNN;
    const float*         rpr  = rpe  + (size_t)sn * KNN * 4;
    const float scale = 0.17677669529663687f;   // 1/sqrt(32)

    // hoist loop-invariant rank-4 weights / biases for this channel
    const float wk0 = Wrk[c], wk1 = Wrk[HD + c], wk2 = Wrk[2 * HD + c], wk3 = Wrk[3 * HD + c];
    const float wv0 = Wrv[c], wv1 = Wrv[HD + c], wv2 = Wrv[2 * HD + c], wv3 = Wrv[3 * HD + c];
    const float bk = brk[c], bv = brv[c];

    for (int k = 0; k < KNN; ++k) {
        int j = idxr[k];
        const float4 r = *(const float4*)&rpr[k * 4];
        float kv = Kx[((size_t)s * N + j) * HD + c]
                 + r.x * wk0 + r.y * wk1 + r.z * wk2 + r.w * wk3 + bk;
        float p = q * kv;
#pragma unroll
        for (int off = 16; off > 0; off >>= 1) p += __shfl_xor(p, off, 32);
        if (d == 0) sc[h][k] = ivr[k] ? -1.0e9f : p * scale;
    }
    __syncthreads();

    float mx = -3.0e38f;
    for (int k = 0; k < KNN; ++k) mx = fmaxf(mx, sc[h][k]);
    float den = 0.0f;
    for (int k = 0; k < KNN; ++k) den += __expf(sc[h][k] - mx);
    float invden = 1.0f / den;

    float o = 0.0f;
    for (int k = 0; k < KNN; ++k) {
        int j = idxr[k];
        const float4 r = *(const float4*)&rpr[k * 4];
        float vv = Vx[((size_t)s * N + j) * HD + c]
                 + r.x * wv0 + r.y * wv1 + r.z * wv2 + r.w * wv3 + bv;
        o += __expf(sc[h][k] - mx) * invden * vv;
    }
    O[(size_t)sn * HD + c] = o;
}

// =====================================================================
// Row LayerNorm (residual already fused into preceding GEMM), optional mask.
// =====================================================================
__global__ __launch_bounds__(256) void ln_kernel(
    const float* __restrict__ X, const float* __restrict__ g,
    const float* __restrict__ b, const unsigned char* __restrict__ valid,
    float* __restrict__ Y)
{
    int row = blockIdx.x, c = threadIdx.x;
    __shared__ float red[HD];
    float v = X[(size_t)row * HD + c];
    red[c] = v; __syncthreads();
    for (int s = HD / 2; s > 0; s >>= 1) { if (c < s) red[c] += red[c + s]; __syncthreads(); }
    float mu = red[0] * (1.0f / HD);
    __syncthreads();
    float dv = v - mu;
    red[c] = dv * dv; __syncthreads();
    for (int s = HD / 2; s > 0; s >>= 1) { if (c < s) red[c] += red[c + s]; __syncthreads(); }
    float var = red[0] * (1.0f / HD);
    float y = dv * rsqrtf(var + 1e-5f) * g[c] + b[c];
    if (valid && !valid[row]) y = 0.0f;
    Y[(size_t)row * HD + c] = y;
}

__global__ void writeout_kernel(const float* __restrict__ x,
                                const unsigned char* __restrict__ valid,
                                float* __restrict__ out_emb,
                                float* __restrict__ out_valid, int M)
{
    int t = blockIdx.x * blockDim.x + threadIdx.x;
    if (t < M * HD) out_emb[t] = x[t];
    if (t < M)      out_valid[t] = valid[t] ? 1.0f : 0.0f;
}

// =====================================================================
// Host side: param pytree walker (JAX sorted-key flatten order) + pipeline
// =====================================================================
struct Lin { const float* b; const float* w; };
static Lin take_lin(const float*& p, int din, int dout) {
    Lin l; l.b = p; p += dout; l.w = p; p += (size_t)din * dout; return l;
}
struct LayerP {
    Lin ff1, ff2;
    const float *ln1_b, *ln1_g, *ln2_b, *ln2_g;
    Lin rpe_k, rpe_v, wk, wo, wq, wv;
};
static LayerP take_layer(const float*& p) {
    LayerP L;
    L.ff1 = take_lin(p, HD, 4 * HD);
    L.ff2 = take_lin(p, 4 * HD, HD);
    L.ln1_b = p; p += HD;  L.ln1_g = p; p += HD;
    L.ln2_b = p; p += HD;  L.ln2_g = p; p += HD;
    L.rpe_k = take_lin(p, 4, HD);
    L.rpe_v = take_lin(p, 4, HD);
    L.wk = take_lin(p, HD, HD);
    L.wo = take_lin(p, HD, HD);
    L.wq = take_lin(p, HD, HD);
    L.wv = take_lin(p, HD, HD);
    return L;
}
struct ClassP { LayerP layers[2]; Lin mlp1, mlp2; };
static ClassP take_class(const float*& p, int din) {
    ClassP c;
    c.layers[0] = take_layer(p);
    c.layers[1] = take_layer(p);
    c.mlp1 = take_lin(p, din, HD);
    c.mlp2 = take_lin(p, HD, HD);
    return c;
}

static void gemm(const float* A, const Lin& L, const float* residual,
                 const unsigned char* mask, float* C, int M, int Kd, int Nd,
                 int relu, hipStream_t stream)
{
    dim3 grid((M + 63) / 64, Nd / 32);
    gemm_kernel<<<grid, 128, 0, stream>>>(A, L.w, L.b, residual, mask, C, M, Kd, Nd, relu);
}

struct Scratch {
    float *x, *q, *k, *v, *o, *t, *t2, *ff, *rpe;
    int *idx; unsigned char *invk;
};

static void run_class(const ClassP& P, const float* attr, const unsigned char* valid,
                      const float* rel_pose, const float* rel_dist,
                      int N, int off, int din,
                      float* out_emb, float* out_valid,
                      const Scratch& W, hipStream_t stream)
{
    int M = NSCENE * N;

    knn_kernel<<<(M + 127) / 128, 128, 0, stream>>>(
        valid, rel_pose, rel_dist, W.idx, W.invk, W.rpe, N, off);

    // embedding MLP
    gemm(attr, P.mlp1, nullptr, nullptr, W.t, M, din, HD, 1, stream);
    gemm(W.t,  P.mlp2, nullptr, valid,   W.x, M, HD,  HD, 1, stream);

    for (int l = 0; l < 2; ++l) {
        const LayerP& L = P.layers[l];
        gemm(W.x, L.wq, nullptr, nullptr, W.q, M, HD, HD, 0, stream);
        gemm(W.x, L.wk, nullptr, nullptr, W.k, M, HD, HD, 0, stream);
        gemm(W.x, L.wv, nullptr, nullptr, W.v, M, HD, HD, 0, stream);

        attn_kernel<<<M, 256, 0, stream>>>(
            W.q, W.k, W.v, W.rpe, W.idx, W.invk,
            L.rpe_k.w, L.rpe_k.b, L.rpe_v.w, L.rpe_v.b, W.o, N);

        gemm(W.o, L.wo, W.x, nullptr, W.t2, M, HD, HD, 0, stream);           // + residual x
        ln_kernel<<<M, 256, 0, stream>>>(W.t2, L.ln1_g, L.ln1_b, nullptr, W.t);

        gemm(W.t,  L.ff1, nullptr, nullptr, W.ff, M, HD,     4 * HD, 1, stream);
        gemm(W.ff, L.ff2, W.t,     nullptr, W.t2, M, 4 * HD, HD,     0, stream); // + residual
        ln_kernel<<<M, 256, 0, stream>>>(W.t2, L.ln2_g, L.ln2_b, valid, W.x);
    }

    writeout_kernel<<<(M * HD + 255) / 256, 256, 0, stream>>>(
        W.x, valid, out_emb, out_valid, M);
}

extern "C" void kernel_launch(void* const* d_in, const int* in_sizes, int n_in,
                              void* d_out, int out_size, void* d_ws, size_t ws_size,
                              hipStream_t stream)
{
    if (n_in < 9) return;
    const unsigned char* agent_valid = (const unsigned char*)d_in[0];
    const float*         agent_attr  = (const float*)d_in[1];
    const unsigned char* map_valid   = (const unsigned char*)d_in[2];
    const float*         map_attr    = (const float*)d_in[3];
    const unsigned char* tl_valid    = (const unsigned char*)d_in[4];
    const float*         tl_attr     = (const float*)d_in[5];
    const float*         rel_pose    = (const float*)d_in[6];
    const float*         rel_dist    = (const float*)d_in[7];
    const float*         pp          = (const float*)d_in[8];

    // JAX tree_flatten: sorted class keys -> agent(68), map(38), tl(12)
    ClassP agentP = take_class(pp, 68);
    ClassP mapP   = take_class(pp, 38);
    ClassP tlP    = take_class(pp, 12);

    // scratch carve (sized for largest class: map, M = 4096)
    const size_t maxM = (size_t)NSCENE * 1024;
    char* base = (char*)d_ws;
    auto takef = [&](size_t nfloats) { float* p = (float*)base; base += nfloats * sizeof(float); return p; };
    Scratch W;
    W.x   = takef(maxM * HD);
    W.q   = takef(maxM * HD);
    W.k   = takef(maxM * HD);
    W.v   = takef(maxM * HD);
    W.o   = takef(maxM * HD);
    W.t   = takef(maxM * HD);
    W.t2  = takef(maxM * HD);
    W.ff  = takef(maxM * 4 * HD);
    W.rpe = takef(maxM * KNN * 4);
    W.idx = (int*)base;            base += maxM * KNN * sizeof(int);
    W.invk = (unsigned char*)base; base += maxM * KNN;

    // output layout: map_emb, map_valid, tl_emb, tl_valid, agent_emb, agent_valid
    float* out = (float*)d_out;
    float* o_map_emb = out;
    float* o_map_v   = o_map_emb + (size_t)NSCENE * 1024 * HD;
    float* o_tl_emb  = o_map_v   + (size_t)NSCENE * 1024;
    float* o_tl_v    = o_tl_emb  + (size_t)NSCENE * 256 * HD;
    float* o_ag_emb  = o_tl_v    + (size_t)NSCENE * 256;
    float* o_ag_v    = o_ag_emb  + (size_t)NSCENE * 64 * HD;

    run_class(mapP,   map_attr,   map_valid,   rel_pose, rel_dist, 1024, 0,    38, o_map_emb, o_map_v, W, stream);
    run_class(tlP,    tl_attr,    tl_valid,    rel_pose, rel_dist, 256,  1024, 12, o_tl_emb,  o_tl_v,  W, stream);
    run_class(agentP, agent_attr, agent_valid, rel_pose, rel_dist, 64,   1280, 68, o_ag_emb,  o_ag_v,  W, stream);
}
